// MoeBase_59313498358382
// MI455X (gfx1250) — matrix operand order repaired and checked by reading, not compiled, not run on hardware
//
#include <hip/hip_runtime.h>
#include <hip/hip_bf16.h>

// ---------------- problem constants (from reference setup) ----------------
#define BB   4
#define TT   2048
#define NT   (BB * TT)       // 8192 tokens
#define DD   1024
#define HH   2048
#define OO   1024
#define EE   8
#define CAP  NT              // max tokens per expert (top-2 of 8 distinct -> <= NT)

// ---------------- GEMM tiling ----------------
#define TM   64              // token rows per block
#define TN   128             // output cols per block
#define KC   32              // K chunk (matches 16x16x32 WMMA)
#define APAD 8               // pad in halfs: row stride 40 halfs = 80B (16B multiple)
#define BPAD 8

typedef __attribute__((ext_vector_type(16))) _Float16 v16h;
typedef __attribute__((ext_vector_type(8)))  _Float16 v8h;
typedef __attribute__((ext_vector_type(8)))  float    v8f;

__device__ __forceinline__ v16h cat8(v8h lo, v8h hi) {
    return __builtin_shufflevector(lo, hi, 0,1,2,3,4,5,6,7,8,9,10,11,12,13,14,15);
}

// ---- gfx1250 async global->LDS copy (ASYNCcnt-tracked), per-lane 16B ----
__device__ __forceinline__ void async_b128(const void* gsrc, const void* ldsdst) {
    // generic LDS pointer: low 32 bits are the wave-relative LDS byte address
    unsigned lds32 = (unsigned)(unsigned long long)ldsdst;
    asm volatile("global_load_async_to_lds_b128 %0, %1, off"
                 :: "v"(lds32), "v"(gsrc)
                 : "memory");
}
__device__ __forceinline__ void async_wait_all() {
    asm volatile("s_wait_asynccnt 0x0" ::: "memory");
}

// ---------------- kernel 0: zero output + counters ----------------
__global__ __launch_bounds__(256) void moe_zero(float* __restrict__ out, int n,
                                                int* __restrict__ counts) {
    int i = blockIdx.x * 256 + threadIdx.x;
    if (i < n) out[i] = 0.0f;
    if (i < EE) counts[i] = 0;
}

// ---------------- kernel 1: x (fp32) -> xh (f16), contiguous ----------------
__global__ __launch_bounds__(256) void moe_cvt_x(const float* __restrict__ x,
                                                 _Float16* __restrict__ xh) {
    int i = (blockIdx.x * 256 + threadIdx.x) * 8;
    if (i >= NT * DD) return;
    const float* src = x + i;
    v8h o;
#pragma unroll
    for (int j = 0; j < 8; ++j) o[j] = (_Float16)src[j];
    *(v8h*)(xh + i) = o;
}

// ---------------- kernel 2: W [E][K][N] fp32 -> Wt [E][N][K] f16 (LDS transpose) ----------------
// grid: (K/32, N/32, EE), 256 threads
__global__ __launch_bounds__(256) void moe_cvt_w(const float* __restrict__ W,
                                                 _Float16* __restrict__ Wt,
                                                 int K, int N) {
    __shared__ _Float16 tile[32][33];
    const int e  = blockIdx.z;
    const int k0 = blockIdx.x * 32;
    const int n0 = blockIdx.y * 32;
    const int t  = threadIdx.x;
    const int c  = t & 31;     // fast index
    const int r  = t >> 5;     // 0..7

#pragma unroll
    for (int i = 0; i < 4; ++i) {
        int k = r + 8 * i;
        tile[k][c] = (_Float16)W[((size_t)e * K + k0 + k) * N + n0 + c];  // coalesced over n
    }
    __syncthreads();
#pragma unroll
    for (int i = 0; i < 4; ++i) {
        int n = r + 8 * i;
        Wt[((size_t)e * N + n0 + n) * K + k0 + c] = tile[c][n];           // coalesced over k
    }
}

// ---------------- kernel 3: gating (softmax + top-2 + L1 norm) ----------------
__global__ __launch_bounds__(256) void moe_gate(const float* __restrict__ x,
                                                const float* __restrict__ Wg,
                                                const float* __restrict__ bg,
                                                int* __restrict__ counts,
                                                int* __restrict__ tokbuf,
                                                float* __restrict__ gatebuf) {
    const int t   = blockIdx.x;
    const int tid = threadIdx.x;

    float acc[EE];
#pragma unroll
    for (int e = 0; e < EE; ++e) acc[e] = 0.0f;

    const float* xp = x + (size_t)t * DD;
    for (int d = tid; d < DD; d += 256) {
        float xv = xp[d];
        const float* wr = Wg + (size_t)d * EE;
#pragma unroll
        for (int e = 0; e < EE; ++e) acc[e] += xv * wr[e];
    }

    __shared__ float red[256][EE];
#pragma unroll
    for (int e = 0; e < EE; ++e) red[tid][e] = acc[e];
    __syncthreads();
    for (int s = 128; s > 0; s >>= 1) {
        if (tid < s) {
#pragma unroll
            for (int e = 0; e < EE; ++e) red[tid][e] += red[tid + s][e];
        }
        __syncthreads();
    }

    if (tid == 0) {
        float l[EE];
#pragma unroll
        for (int e = 0; e < EE; ++e) l[e] = red[0][e] + bg[e];
        // top-2, lowest-index tie break (matches jax.lax.top_k)
        int i1 = 0;
#pragma unroll
        for (int e = 1; e < EE; ++e) if (l[e] > l[i1]) i1 = e;
        int i2 = (i1 == 0) ? 1 : 0;
#pragma unroll
        for (int e = 0; e < EE; ++e) if (e != i1 && l[e] > l[i2]) i2 = e;
        // softmax denominator cancels under top-2 L1 renorm
        float e2  = __expf(l[i2] - l[i1]);
        float inv = 1.0f / (1.0f + e2);

        int p1 = atomicAdd(&counts[i1], 1);
        tokbuf[i1 * CAP + p1]  = t;
        gatebuf[i1 * CAP + p1] = inv;
        int p2 = atomicAdd(&counts[i2], 1);
        tokbuf[i2 * CAP + p2]  = t;
        gatebuf[i2 * CAP + p2] = e2 * inv;
    }
}

// ---------------- kernel 4: prefix offsets over 8 experts ----------------
__global__ void moe_offsets(const int* __restrict__ counts, int* __restrict__ offsets) {
    if (threadIdx.x == 0 && blockIdx.x == 0) {
        int o = 0;
#pragma unroll
        for (int e = 0; e < EE; ++e) { offsets[e] = o; o += counts[e]; }
    }
}

// ---------------- kernel 5: GEMM1  h = relu(xh @ W1[e] + b1[e]) -> f16 ws ----------------
// grid: (CAP/TM, HH/TN, EE), 256 threads (8 waves). A gathered per token via async.
__global__ __launch_bounds__(256) void moe_gemm1(const _Float16* __restrict__ xh,
                                                 const _Float16* __restrict__ Wt1,
                                                 const float* __restrict__ b1,
                                                 const int* __restrict__ counts,
                                                 const int* __restrict__ offsets,
                                                 const int* __restrict__ tokbuf,
                                                 _Float16* __restrict__ h_ws) {
    const int e   = blockIdx.z;
    const int cnt = counts[e];
    const int m0g = blockIdx.x * TM;
    if (m0g >= cnt) return;
    const int n0g   = blockIdx.y * TN;   // over H
    const int hbase = offsets[e];

    __shared__ __align__(16) _Float16 As[2][TM][KC + APAD];
    __shared__ __align__(16) _Float16 Bs[2][TN][KC + BPAD];
    __shared__ int toks[TM];

    const int tid  = threadIdx.x;
    const int lane = tid & 31;
    const int wave = tid >> 5;
    const int mw   = (wave & 3) * 16;
    const int nw   = (wave >> 2) * 64;

    if (tid < TM) {
        int p = m0g + tid;
        toks[tid] = (p < cnt) ? tokbuf[e * CAP + p] : 0;
    }
    __syncthreads();

    // per-thread staging coordinates
    const int arowS = tid >> 2;                 // 0..63
    const int akk   = (tid & 3) * 8;            // 0,8,16,24
    const _Float16* aSrc = xh + (size_t)toks[arowS] * DD + akk;
    const int c1  = tid, c2 = tid + 256;        // two B chunks of 8 halfs
    const int bn1 = c1 >> 2, bk1 = (c1 & 3) * 8;
    const int bn2 = c2 >> 2, bk2 = (c2 & 3) * 8;
    const _Float16* bSrc1 = Wt1 + ((size_t)e * HH + n0g + bn1) * DD + bk1;
    const _Float16* bSrc2 = Wt1 + ((size_t)e * HH + n0g + bn2) * DD + bk2;

    const v8f vzero = {0.f,0.f,0.f,0.f,0.f,0.f,0.f,0.f};
    v8f acc[4];
#pragma unroll
    for (int j = 0; j < 4; ++j) acc[j] = vzero;

    const int arow = mw + (lane & 15);
    const int ak   = (lane < 16) ? 0 : 8;
    const int bk   = (lane < 16) ? 0 : 8;

    auto issue = [&](int k0, int buf) {
        async_b128(aSrc  + k0, &As[buf][arowS][akk]);
        async_b128(bSrc1 + k0, &Bs[buf][bn1][bk1]);
        async_b128(bSrc2 + k0, &Bs[buf][bn2][bk2]);
    };
    auto compute = [&](int buf) {
        v8h a_lo = *(const v8h*)&As[buf][arow][ak];
        v8h a_hi = *(const v8h*)&As[buf][arow][16 + ak];
        v16h afrag = cat8(a_lo, a_hi);
#pragma unroll
        for (int j = 0; j < 4; ++j) {
            const int bn = nw + j * 16 + (lane & 15);
            v8h b_lo = *(const v8h*)&Bs[buf][bn][bk];
            v8h b_hi = *(const v8h*)&Bs[buf][bn][bk + 16];
            acc[j] = __builtin_amdgcn_wmma_f32_16x16x32_f16(
                false, afrag, false, cat8(b_lo, b_hi), (short)0, acc[j], false, false);
        }
    };

    const int nch = DD / KC;                    // 32 chunks (even)
    issue(0, 0);
#pragma unroll 1
    for (int i = 0; i < nch; i += 2) {
        async_wait_all();                       // chunk i landed (mine)
        __syncthreads();                        // everyone's landed; buf1 free
        issue((i + 1) * KC, 1);                 // i+1 <= nch-1 always
        compute(0);
        async_wait_all();                       // chunk i+1 landed
        __syncthreads();                        // buf0 free
        if (i + 2 < nch) issue((i + 2) * KC, 0);
        compute(1);
    }

    // ---- epilogue: bias + relu, store h as f16 (masked by count) ----
#pragma unroll
    for (int j = 0; j < 4; ++j) {
        const int col  = n0g + nw + j * 16 + (lane & 15);
        const float bv = b1[e * HH + col];
#pragma unroll
        for (int r = 0; r < 8; ++r) {
            const int m = mw + r + ((lane >> 4) << 3);  // C layout: M = r + 8*(lane/16)
            const int p = m0g + m;
            if (p < cnt) {
                float v = acc[j][r] + bv;
                v = v > 0.0f ? v : 0.0f;
                h_ws[(size_t)(hbase + p) * HH + col] = (_Float16)v;
            }
        }
    }
}

// ---------------- kernel 6: GEMM2  out += g * (h @ W2[e] + b2[e]) ----------------
// grid: (CAP/TM, OO/TN, EE), 256 threads
__global__ __launch_bounds__(256) void moe_gemm2(const _Float16* __restrict__ h_ws,
                                                 const _Float16* __restrict__ Wt2,
                                                 const float* __restrict__ b2,
                                                 const int* __restrict__ counts,
                                                 const int* __restrict__ offsets,
                                                 const int* __restrict__ tokbuf,
                                                 const float* __restrict__ gatebuf,
                                                 float* __restrict__ out) {
    const int e   = blockIdx.z;
    const int cnt = counts[e];
    const int m0g = blockIdx.x * TM;
    if (m0g >= cnt) return;
    const int n0g   = blockIdx.y * TN;   // over O
    const int hbase = offsets[e];

    __shared__ __align__(16) _Float16 As[2][TM][KC + APAD];
    __shared__ __align__(16) _Float16 Bs[2][TN][KC + BPAD];
    __shared__ int   toks[TM];
    __shared__ float gbuf[TM];

    const int tid  = threadIdx.x;
    const int lane = tid & 31;
    const int wave = tid >> 5;
    const int mw   = (wave & 3) * 16;
    const int nw   = (wave >> 2) * 64;

    if (tid < TM) {
        int p = m0g + tid;
        toks[tid] = (p < cnt) ? tokbuf[e * CAP + p]  : 0;
        gbuf[tid] = (p < cnt) ? gatebuf[e * CAP + p] : 0.0f;
    }
    __syncthreads();

    const int arowS = tid >> 2;
    const int akk   = (tid & 3) * 8;
    const int prS   = (m0g + arowS < cnt) ? (m0g + arowS) : 0;  // clamp, masked at store
    const _Float16* aSrc = h_ws + (size_t)(hbase + prS) * HH + akk;
    const int c1  = tid, c2 = tid + 256;
    const int bn1 = c1 >> 2, bk1 = (c1 & 3) * 8;
    const int bn2 = c2 >> 2, bk2 = (c2 & 3) * 8;
    const _Float16* bSrc1 = Wt2 + ((size_t)e * OO + n0g + bn1) * HH + bk1;
    const _Float16* bSrc2 = Wt2 + ((size_t)e * OO + n0g + bn2) * HH + bk2;

    const v8f vzero = {0.f,0.f,0.f,0.f,0.f,0.f,0.f,0.f};
    v8f acc[4];
#pragma unroll
    for (int j = 0; j < 4; ++j) acc[j] = vzero;

    const int arow = mw + (lane & 15);
    const int ak   = (lane < 16) ? 0 : 8;
    const int bk   = (lane < 16) ? 0 : 8;

    auto issue = [&](int k0, int buf) {
        async_b128(aSrc  + k0, &As[buf][arowS][akk]);
        async_b128(bSrc1 + k0, &Bs[buf][bn1][bk1]);
        async_b128(bSrc2 + k0, &Bs[buf][bn2][bk2]);
    };
    auto compute = [&](int buf) {
        v8h a_lo = *(const v8h*)&As[buf][arow][ak];
        v8h a_hi = *(const v8h*)&As[buf][arow][16 + ak];
        v16h afrag = cat8(a_lo, a_hi);
#pragma unroll
        for (int j = 0; j < 4; ++j) {
            const int bn = nw + j * 16 + (lane & 15);
            v8h b_lo = *(const v8h*)&Bs[buf][bn][bk];
            v8h b_hi = *(const v8h*)&Bs[buf][bn][bk + 16];
            acc[j] = __builtin_amdgcn_wmma_f32_16x16x32_f16(
                false, afrag, false, cat8(b_lo, b_hi), (short)0, acc[j], false, false);
        }
    };

    const int nch = HH / KC;                    // 64 chunks (even)
    issue(0, 0);
#pragma unroll 1
    for (int i = 0; i < nch; i += 2) {
        async_wait_all();
        __syncthreads();
        issue((i + 1) * KC, 1);
        compute(0);
        async_wait_all();
        __syncthreads();
        if (i + 2 < nch) issue((i + 2) * KC, 0);
        compute(1);
    }

    // ---- epilogue: bias, gate-scale, atomic combine (2 commutative adds -> deterministic) ----
#pragma unroll
    for (int j = 0; j < 4; ++j) {
        const int col  = n0g + nw + j * 16 + (lane & 15);
        const float bv = b2[e * OO + col];
#pragma unroll
        for (int r = 0; r < 8; ++r) {
            const int m = mw + r + ((lane >> 4) << 3);
            const int p = m0g + m;
            if (p < cnt) {
                float v = (acc[j][r] + bv) * gbuf[m];
                atomicAdd(&out[(size_t)toks[m] * OO + col], v);
            }
        }
    }
}

// ---------------- launcher ----------------
extern "C" void kernel_launch(void* const* d_in, const int* in_sizes, int n_in,
                              void* d_out, int out_size, void* d_ws, size_t ws_size,
                              hipStream_t stream) {
    const float* x  = (const float*)d_in[0];
    // d_in[1] = num_experts_per_token (== 2, baked into top-2 gating)
    const float* Wg = (const float*)d_in[2];
    const float* bg = (const float*)d_in[3];
    const float* W1 = (const float*)d_in[4];
    const float* b1 = (const float*)d_in[5];
    const float* W2 = (const float*)d_in[6];
    const float* b2 = (const float*)d_in[7];
    float* out = (float*)d_out;

    // workspace carve-up (16B aligned regions):
    //  counts(8 i32) offsets(8 i32) tokbuf(EE*CAP i32) gatebuf(EE*CAP f32)
    //  xh  : NT*DD        f16   (16 MB)
    //  Wt1 : EE*HH*DD     f16   (32 MB)   column-major [e][h][d]
    //  Wt2 : EE*OO*HH     f16   (32 MB)   column-major [e][o][h]
    //  h_ws: (2*NT)*HH    f16   (67 MB)
    int*      counts  = (int*)d_ws;
    int*      offsets = counts + 8;
    int*      tokbuf  = counts + 16;
    float*    gatebuf = (float*)(tokbuf + EE * CAP);
    _Float16* xh      = (_Float16*)(gatebuf + EE * CAP);
    _Float16* Wt1     = xh  + (size_t)NT * DD;
    _Float16* Wt2     = Wt1 + (size_t)EE * HH * DD;
    _Float16* h_ws    = Wt2 + (size_t)EE * OO * HH;

    const int nOut = NT * OO;
    moe_zero<<<(nOut + 255) / 256, 256, 0, stream>>>(out, nOut, counts);
    moe_cvt_x<<<(NT * DD) / (256 * 8), 256, 0, stream>>>(x, xh);
    moe_cvt_w<<<dim3(DD / 32, HH / 32, EE), 256, 0, stream>>>(W1, Wt1, DD, HH);
    moe_cvt_w<<<dim3(HH / 32, OO / 32, EE), 256, 0, stream>>>(W2, Wt2, HH, OO);
    moe_gate<<<NT, 256, 0, stream>>>(x, Wg, bg, counts, tokbuf, gatebuf);
    moe_offsets<<<1, 32, 0, stream>>>(counts, offsets);
    moe_gemm1<<<dim3(CAP / TM, HH / TN, EE), 256, 0, stream>>>(
        xh, Wt1, b1, counts, offsets, tokbuf, h_ws);
    moe_gemm2<<<dim3(CAP / TM, OO / TN, EE), 256, 0, stream>>>(
        h_ws, Wt2, b2, counts, offsets, tokbuf, gatebuf, out);
}
